// STWDNN_44513041056056
// MI455X (gfx1250) — matrix operand order repaired and checked by reading, not compiled, not run on hardware
//
#include <hip/hip_runtime.h>
#include <math.h>

// ---------- types / helpers ----------
typedef __attribute__((ext_vector_type(16))) __bf16    v16bf;
typedef __attribute__((ext_vector_type(8)))  float     v8f;
typedef __attribute__((ext_vector_type(8)))  unsigned  v8u;
typedef __attribute__((ext_vector_type(4)))  unsigned  v4u;

#define DEV __device__ __forceinline__

union FragU {
  v8u v;
  v4u q[2];
  unsigned u[8];
};

DEV unsigned short f2bf_bits(float f) {
  unsigned u = __builtin_bit_cast(unsigned, f);
  u = u + 0x7FFFu + ((u >> 16) & 1u);
  return (unsigned short)(u >> 16);
}
DEV float bf2f_bits(unsigned s) {
  unsigned u = s << 16;
  return __builtin_bit_cast(float, u);
}

DEV v8f wmma_bf(const FragU& a, const FragU& b, v8f c) {
  return __builtin_amdgcn_wmma_f32_16x16x32_bf16(
      false, __builtin_bit_cast(v16bf, a.v), false, __builtin_bit_cast(v16bf, b.v),
      (short)0, c, false, false);
}

// block = 256 threads (8 wave32)
DEV float blockReduceSum(float v, float* sm) {
  #pragma unroll
  for (int off = 16; off; off >>= 1) v += __shfl_xor(v, off);
  __syncthreads();
  if ((threadIdx.x & 31) == 0) sm[threadIdx.x >> 5] = v;
  __syncthreads();
  if (threadIdx.x == 0) { float r = 0.f; for (int i = 0; i < 8; ++i) r += sm[i]; sm[0] = r; }
  __syncthreads();
  float out = sm[0];
  __syncthreads();
  return out;
}
DEV float blockReduceMax(float v, float* sm) {
  #pragma unroll
  for (int off = 16; off; off >>= 1) v = fmaxf(v, __shfl_xor(v, off));
  __syncthreads();
  if ((threadIdx.x & 31) == 0) sm[threadIdx.x >> 5] = v;
  __syncthreads();
  if (threadIdx.x == 0) { float r = sm[0]; for (int i = 1; i < 8; ++i) r = fmaxf(r, sm[i]); sm[0] = r; }
  __syncthreads();
  float out = sm[0];
  __syncthreads();
  return out;
}
DEV float blockReduceMin(float v, float* sm) {
  #pragma unroll
  for (int off = 16; off; off >>= 1) v = fminf(v, __shfl_xor(v, off));
  __syncthreads();
  if ((threadIdx.x & 31) == 0) sm[threadIdx.x >> 5] = v;
  __syncthreads();
  if (threadIdx.x == 0) { float r = sm[0]; for (int i = 1; i < 8; ++i) r = fminf(r, sm[i]); sm[0] = r; }
  __syncthreads();
  float out = sm[0];
  __syncthreads();
  return out;
}

// ---------- misc ----------
__global__ void zero_f32(float* p, int n) {
  int i = blockIdx.x * 256 + threadIdx.x;
  if (i < n) p[i] = 0.f;
}

// pack f32 weights (M x Kl) -> bf16 pairs (M x Kp/2 uints), zero-padded to Kp
__global__ void wprep(const float* __restrict__ W, unsigned* __restrict__ Wbf,
                      int M, int Kl, int Kp) {
  int half = Kp >> 1;
  int i = blockIdx.x * 256 + threadIdx.x;
  if (i >= M * half) return;
  int m = i / half, j = i - m * half;
  int k = 2 * j;
  unsigned lo = (k     < Kl) ? f2bf_bits(W[(size_t)m * Kl + k])     : 0u;
  unsigned hi = (k + 1 < Kl) ? f2bf_bits(W[(size_t)m * Kl + k + 1]) : 0u;
  Wbf[i] = lo | (hi << 16);
}

// stat slot: [0..C) sum, [256..) sumsq, [512..) scale, [768..) shift
__global__ void bn_finalize(float* stat, const float* g, const float* beta, int C,
                            float eps, float cnt_const, const float* cnt_ptr) {
  int c = blockIdx.x * 256 + threadIdx.x;
  if (c >= C) return;
  float n = cnt_ptr ? cnt_ptr[0] : cnt_const;
  float mean = stat[c] / n;
  float var  = stat[256 + c] / n - mean * mean;
  float inv  = rsqrtf(fmaxf(var, 0.f) + eps);
  float sc   = g[c] * inv;
  stat[512 + c] = sc;
  stat[768 + c] = beta[c] - mean * sc;
}

// ---------- WMMA bf16 GEMM: Y[M][N] = Wbf(M x Kp bf16) * T(X[Kl][N]) + bias ----------
// Block tile 32M x 128N, 8 waves, each wave: one 16M tile x two 16N tiles.
// IN_MODE: 0 raw, 1 affine, 2 relu(affine). NFULL: N multiple of 128. KFULL: Kl==Kp.
template<int IN_MODE, bool NFULL, bool KFULL>
__global__ __launch_bounds__(256) void gemm_bn_t(
    const unsigned* __restrict__ Wbf, const float* __restrict__ bias,
    const unsigned short* __restrict__ X,
    const float* __restrict__ inS, const float* __restrict__ inB,
    unsigned short* __restrict__ Y, float* __restrict__ stat,
    int M, int N, int Kl, int Kp)
{
  __shared__ unsigned Xs[128][17];   // packed bf16 pairs: element(n, 2j|2j+1) at Xs[n][j]
  const int tid = threadIdx.x;
  const int lane = tid & 31, wv = tid >> 5;
  const int hf = lane >> 4, l15 = lane & 15;
  const int mtile = wv & 1, ntile = wv >> 1;
  const int nb = blockIdx.x * 128, mb = blockIdx.y * 32;
  const int m0 = mb + mtile * 16 + l15;
  const int Kq = Kp >> 3;            // row stride in uint4 units

  v8f acc0 = {0.f,0.f,0.f,0.f,0.f,0.f,0.f,0.f};
  v8f acc1 = {0.f,0.f,0.f,0.f,0.f,0.f,0.f,0.f};

  for (int k0 = 0; k0 < Kp; k0 += 32) {
    __syncthreads();
    {
      const int kk = tid >> 3;           // 0..31
      const int cg = (tid & 7) * 16;     // 16 columns per thread
      const int kg = k0 + kk;
      bool kok = KFULL ? true : (kg < Kl);
      float sc = 1.f, sh = 0.f;
      if (IN_MODE && kok) { sc = inS[kg]; sh = inB[kg]; }
      unsigned short* dst = (unsigned short*)&Xs[0][0];
      if (NFULL && kok) {
        const v4u* src = (const v4u*)(X + (size_t)kg * N + nb + cg);
        FragU xin;
        xin.q[0] = src[0];
        xin.q[1] = src[1];
        #pragma unroll
        for (int i = 0; i < 8; ++i) {
          unsigned uu = xin.u[i];
          if (IN_MODE) {
            float lo = bf2f_bits(uu & 0xFFFFu) * sc + sh;
            float hi = bf2f_bits(uu >> 16)     * sc + sh;
            if (IN_MODE == 2) { lo = fmaxf(lo, 0.f); hi = fmaxf(hi, 0.f); }
            dst[(cg + 2 * i)     * 34 + kk] = f2bf_bits(lo);
            dst[(cg + 2 * i + 1) * 34 + kk] = f2bf_bits(hi);
          } else {
            dst[(cg + 2 * i)     * 34 + kk] = (unsigned short)(uu & 0xFFFFu);
            dst[(cg + 2 * i + 1) * 34 + kk] = (unsigned short)(uu >> 16);
          }
        }
      } else {
        #pragma unroll
        for (int i = 0; i < 16; ++i) {
          const int n = nb + cg + i;
          float v = 0.f;
          if (kok && (NFULL || n < N)) {
            v = bf2f_bits((unsigned)X[(size_t)kg * N + n]);
            if (IN_MODE) { v = v * sc + sh; if (IN_MODE == 2) v = fmaxf(v, 0.f); }
          }
          dst[(cg + i) * 34 + kk] = f2bf_bits(v);
        }
      }
    }
    __syncthreads();
    FragU au, b0, b1;
    {
      const v4u* Wq = (const v4u*)Wbf;
      const int base = m0 * Kq + (k0 >> 3) + 2 * hf;
      au.q[0] = Wq[base];
      au.q[1] = Wq[base + 1];
    }
    const int nn0 = ntile * 16 + l15;
    #pragma unroll
    for (int r = 0; r < 8; ++r) {
      b0.u[r] = Xs[nn0][hf * 8 + r];
      b1.u[r] = Xs[nn0 + 64][hf * 8 + r];
    }
    acc0 = wmma_bf(au, b0, acc0);
    acc1 = wmma_bf(au, b1, acc1);
  }

  const int n0 = nb + ntile * 16 + l15;
  const int n1 = n0 + 64;
  const bool ok0 = NFULL || (n0 < N);
  const bool ok1 = NFULL || (n1 < N);
  #pragma unroll
  for (int r = 0; r < 8; ++r) {
    const int m = mb + mtile * 16 + hf * 8 + r;
    const float bs = bias[m];
    const float y0 = acc0[r] + bs;
    const float y1v = acc1[r] + bs;
    if (ok0) Y[(size_t)m * N + n0] = f2bf_bits(y0);
    if (ok1) Y[(size_t)m * N + n1] = f2bf_bits(y1v);
    float s  = (ok0 ? y0 : 0.f)       + (ok1 ? y1v : 0.f);
    float s2 = (ok0 ? y0 * y0 : 0.f)  + (ok1 ? y1v * y1v : 0.f);
    #pragma unroll
    for (int off = 1; off < 16; off <<= 1) {
      s  += __shfl_xor(s, off);
      s2 += __shfl_xor(s2, off);
    }
    if (l15 == 0) { atomicAdd(&stat[m], s); atomicAdd(&stat[256 + m], s2); }
  }
}

// ---------- implicit-GEMM 3x3 conv (WMMA), masked-BN fused ----------
// IN_MODE: 0 raw; 1 mask*(x*sc+sh); 2 mask*relu(x*sc+sh). NFULL: N multiple of 128.
template<int IN_MODE, bool NFULL>
__global__ __launch_bounds__(256) void conv_gemm_t(
    const unsigned* __restrict__ Wbf, const float* __restrict__ bias,
    const float* __restrict__ Xin,
    const float* __restrict__ inS, const float* __restrict__ inB,
    const float* __restrict__ msk,
    float* __restrict__ Y, float* __restrict__ stat,
    int M, int Ci, int N, int Wd, int Kp)
{
  __shared__ unsigned Xs[128][17];
  const int tid = threadIdx.x;
  const int lane = tid & 31, wv = tid >> 5;
  const int hf = lane >> 4, l15 = lane & 15;
  const int mtile = wv & 1, ntile = wv >> 1;
  const int nb = blockIdx.x * 128, mb = blockIdx.y * 32;
  const int HW = Wd * Wd;
  const int K = Ci * 9;
  const int m0 = mb + mtile * 16 + l15;
  const int Kq = Kp >> 3;

  v8f acc0 = {0.f,0.f,0.f,0.f,0.f,0.f,0.f,0.f};
  v8f acc1 = {0.f,0.f,0.f,0.f,0.f,0.f,0.f,0.f};

  for (int k0 = 0; k0 < Kp; k0 += 32) {
    __syncthreads();
    {
      const int kk = tid >> 3;
      const int cg = (tid & 7) * 16;
      const int kg = k0 + kk;
      const bool kok = kg < K;
      int ci = 0, ky = 0, kx = 0;
      float sc = 1.f, sh = 0.f;
      if (kok) {
        ci = kg / 9;
        int r = kg - ci * 9;
        ky = r / 3 - 1;
        kx = r - (r / 3) * 3 - 1;
        if (IN_MODE) { sc = inS[ci]; sh = inB[ci]; }
      }
      unsigned short* dst = (unsigned short*)&Xs[0][0];
      #pragma unroll
      for (int i = 0; i < 16; ++i) {
        const int n = nb + cg + i;
        float v = 0.f;
        if (kok && (NFULL || n < N)) {
          const int b = n / HW;
          const int p = n - b * HW;
          const int yy = p / Wd + ky;
          const int xx = p - (p / Wd) * Wd + kx;
          if (yy >= 0 && yy < Wd && xx >= 0 && xx < Wd) {
            const int q = b * HW + yy * Wd + xx;
            v = Xin[(size_t)ci * N + q];
            if (IN_MODE) {
              v = v * sc + sh;
              if (IN_MODE == 2) v = fmaxf(v, 0.f);
              v *= msk[q];
            }
          }
        }
        dst[(cg + i) * 34 + kk] = f2bf_bits(v);
      }
    }
    __syncthreads();
    FragU au, b0, b1;
    {
      const v4u* Wq = (const v4u*)Wbf;
      const int base = m0 * Kq + (k0 >> 3) + 2 * hf;
      au.q[0] = Wq[base];
      au.q[1] = Wq[base + 1];
    }
    const int nn0 = ntile * 16 + l15;
    #pragma unroll
    for (int r = 0; r < 8; ++r) {
      b0.u[r] = Xs[nn0][hf * 8 + r];
      b1.u[r] = Xs[nn0 + 64][hf * 8 + r];
    }
    acc0 = wmma_bf(au, b0, acc0);
    acc1 = wmma_bf(au, b1, acc1);
  }

  const int n0 = nb + ntile * 16 + l15;
  const int n1 = n0 + 64;
  const bool ok0 = NFULL || (n0 < N);
  const bool ok1 = NFULL || (n1 < N);
  const float m0v = ok0 ? msk[n0] : 0.f;
  const float m1v = ok1 ? msk[n1] : 0.f;
  #pragma unroll
  for (int r = 0; r < 8; ++r) {
    const int m = mb + mtile * 16 + hf * 8 + r;
    const float bs = bias[m];
    const float y0 = (acc0[r] + bs) * m0v;
    const float y1v = (acc1[r] + bs) * m1v;
    if (ok0) Y[(size_t)m * N + n0] = y0;
    if (ok1) Y[(size_t)m * N + n1] = y1v;
    float s  = y0 + y1v;
    float s2 = y0 * y0 + y1v * y1v;
    #pragma unroll
    for (int off = 1; off < 16; off <<= 1) {
      s  += __shfl_xor(s, off);
      s2 += __shfl_xor(s2, off);
    }
    if (l15 == 0) { atomicAdd(&stat[m], s); atomicAdd(&stat[256 + m], s2); }
  }
}

// ---------- point branch ----------
__global__ void pf_stats(const float* pf, float* stat) {
  __shared__ float sm[8];
  int c = blockIdx.x, t = threadIdx.x;
  float v = (t < 200) ? pf[t * 14 + c] : 0.f;
  float s  = blockReduceSum(v, sm);
  float s2 = blockReduceSum(v * v, sm);
  if (t == 0) { stat[c] = s; stat[256 + c] = s2; }
}

__global__ void pf_apply(const float* pf, const float* stat, float* pfbn, unsigned short* Xpf) {
  int i = blockIdx.x * 256 + threadIdx.x;
  if (i >= 200 * 14) return;
  int b = i / 14, c = i - b * 14;
  float v = pf[i] * stat[512 + c] + stat[768 + c];
  pfbn[i] = v;
  Xpf[c * 200 + b] = f2bf_bits(v);
}

// ---------- station branch ----------
__global__ void dstf_build(const float* sfeat, const float* spm, const float* pfbn,
                           unsigned short* act, float* stat) {
  __shared__ float sm[8];
  int n = blockIdx.x * 256 + threadIdx.x;   // N = 204800 exactly
  int b = n >> 10;
  for (int c = 0; c < 15; ++c) {
    float v = (c < 14) ? (sfeat[(size_t)n * 14 + c] - pfbn[b * 14 + c]) : spm[n];
    act[(size_t)c * 204800 + n] = f2bf_bits(v);
    float s  = blockReduceSum(v, sm);
    float s2 = blockReduceSum(v * v, sm);
    if (threadIdx.x == 0) { atomicAdd(&stat[c], s); atomicAdd(&stat[256 + c], s2); }
  }
}

__global__ void smask_sum(const float* smask, float* dsum) {
  __shared__ float sm[8];
  int b = blockIdx.x;
  float v = 0.f;
  for (int s = threadIdx.x; s < 1024; s += 256) v += smask[b * 1024 + s];
  float t = blockReduceSum(v, sm);
  if (threadIdx.x == 0) dsum[b] = t;
}

__global__ void station_reduce(const unsigned short* act, const float* stat,
                               const float* smask, const float* dsum, float* outr) {
  __shared__ float sm[8];
  int bc = blockIdx.x, b = bc >> 8, c = bc & 255;
  float sc = stat[512 + c], sh = stat[768 + c];
  float a = 0.f;
  for (int s = threadIdx.x; s < 1024; s += 256) {
    float v = bf2f_bits((unsigned)act[(size_t)c * 204800 + b * 1024 + s]);
    a += fmaxf(v * sc + sh, 0.f) * smask[b * 1024 + s];
  }
  float t = blockReduceSum(a, sm);
  if (threadIdx.x == 0) outr[b * 256 + c] = t / (dsum[b] + 1e-7f);
}

// ---------- point head ----------
__global__ void pf1_head(const unsigned short* act, const float* stat, float* pf1z, float* at2) {
  __shared__ float sm[8];
  int b = blockIdx.x, c = threadIdx.x;
  float v = bf2f_bits((unsigned)act[c * 200 + b]);
  float z = fmaxf(v * stat[512 + c] + stat[768 + c], 0.f);
  pf1z[b * 256 + c] = z;
  float s  = blockReduceSum(z, sm);
  float mx = blockReduceMax(z, sm);
  float mn = blockReduceMin(z, sm);
  float sq = blockReduceSum(z * z, sm);
  if (c == 0) {
    at2[b * 4 + 0] = s / 256.f;
    at2[b * 4 + 1] = mx;
    at2[b * 4 + 2] = mn;
    at2[b * 4 + 3] = sqrtf(fmaxf((sq - s * s / 256.f) / 255.f, 0.f));
  }
}

// ---------- region branch ----------
__global__ void mask_sum(const float* rm, float* nsum) {
  __shared__ float sm[8];
  int b = blockIdx.x;
  float t = blockReduceSum(rm[b * 256 + threadIdx.x], sm);
  if (threadIdx.x == 0) atomicAdd(nsum, t);
}

__global__ void x0_build(const float* rfeat, const float* rm, float* x0, float* stat) {
  __shared__ float sm[8];
  int b = blockIdx.x, p = threadIdx.x;
  float m = rm[b * 256 + p];
  for (int c = 0; c < 11; ++c) {
    float raw = rfeat[(size_t)(b * 256 + p) * 11 + c];
    float ctr = rfeat[(size_t)(b * 256 + 136) * 11 + c];  // (8,8)
    float v = (raw - ctr) * m;
    x0[(size_t)c * 51200 + b * 256 + p] = v;
    float s  = blockReduceSum(v, sm);
    float sq = blockReduceSum(v * v, sm);
    if (p == 0) { atomicAdd(&stat[c], s); atomicAdd(&stat[256 + c], sq); }
  }
}

__global__ void pool_mask(const float* rm, float* den1, float* maxm, float* nsum2) {
  int i = blockIdx.x * 256 + threadIdx.x;
  if (i >= 1800) return;
  int b = i / 9, p = i - b * 9, py = p / 3, px = p - py * 3;
  float s = 0.f, mx = 0.f;
  for (int dy = 0; dy < 8; ++dy)
    for (int dx = 0; dx < 8; ++dx) {
      float m = rm[b * 256 + (py * 4 + dy) * 16 + (px * 4 + dx)];
      s += m; mx = fmaxf(mx, m);
    }
  den1[i] = s + 64e-7f;
  maxm[i] = mx;
  atomicAdd(nsum2, mx);
}

__global__ void pool_feat(const float* y2, const float* rm, const float* stat,
                          const float* den1, const float* maxm, float* avgrf1) {
  int i = blockIdx.x * 256 + threadIdx.x;
  if (i >= 256 * 1800) return;
  int c = i / 1800, bp = i - c * 1800, b = bp / 9, p = bp - b * 9;
  int py = p / 3, px = p - py * 3;
  float sc = stat[512 + c], sh = stat[768 + c];
  float num = 0.f;
  for (int dy = 0; dy < 8; ++dy)
    for (int dx = 0; dx < 8; ++dx) {
      int q = (py * 4 + dy) * 16 + (px * 4 + dx);
      float m = rm[b * 256 + q];
      float v = y2[(size_t)c * 51200 + b * 256 + q];
      num += m * fmaxf(v * sc + sh, 0.f);
    }
  avgrf1[(size_t)c * 1800 + bp] = num / den1[bp] * maxm[bp];
}

__global__ void final_pool(const float* y4, const float* stat, const float* maxm, float* avgrf2) {
  int i = blockIdx.x * 256 + threadIdx.x;
  if (i >= 51200) return;
  int b = i / 256, c = i - b * 256;
  float sc = stat[512 + c], sh = stat[768 + c];
  float num = 0.f, den = 0.f;
  for (int p = 0; p < 9; ++p) {
    float m = maxm[b * 9 + p];
    num += m * fmaxf(y4[(size_t)c * 1800 + b * 9 + p] * sc + sh, 0.f);
    den += m;
  }
  avgrf2[i] = num / (den + 9e-7f);
}

// ---------- final head ----------
__global__ void combine(const float* pf1z, const float* at2, const float* cw, const float* cb,
                        const float* dstf_red, const float* avgrf2, unsigned short* allf) {
  int i = blockIdx.x * 256 + threadIdx.x;
  if (i >= 51200) return;
  int b = i / 256, c = i - b * 256;
  float w = cb[c];
  #pragma unroll
  for (int j = 0; j < 4; ++j) w += at2[b * 4 + j] * cw[c * 4 + j];
  float v = pf1z[i] + w * (dstf_red[i] + avgrf2[i]);
  allf[c * 200 + b] = f2bf_bits(v);
}

__global__ void out_head(const unsigned short* h5, const float* stat,
                         const float* w2, const float* b2, float* out) {
  __shared__ float sm[8];
  int b = blockIdx.x, c = threadIdx.x;
  float v = bf2f_bits((unsigned)h5[c * 200 + b]);
  float z = fmaxf(v * stat[512 + c] + stat[768 + c], 0.f);
  float a = blockReduceSum(z * w2[c], sm);
  if (c == 0) out[b] = 1.f / (1.f + expf(-(a + b2[0])));
}

// ---------- host ----------
extern "C" void kernel_launch(void* const* d_in, const int* in_sizes, int n_in,
                              void* d_out, int out_size, void* d_ws, size_t ws_size,
                              hipStream_t stream) {
  const float* pointfeature  = (const float*)d_in[0];
  const float* stationfeat   = (const float*)d_in[1];
  const float* stationpm     = (const float*)d_in[2];
  const float* stationmask   = (const float*)d_in[3];
  const float* regionfeat    = (const float*)d_in[4];
  const float* regionmask    = (const float*)d_in[5];
  const float* bnraw1_beta = (const float*)d_in[6],  *bnraw1_g = (const float*)d_in[7];
  const float* bnraw2_beta = (const float*)d_in[8],  *bnraw2_g = (const float*)d_in[9];
  const float* cfe_W = (const float*)d_in[10], *cfe_b = (const float*)d_in[11];
  const float* ofe_W1 = (const float*)d_in[12], *ofe_W2 = (const float*)d_in[13];
  const float* ofe_b1 = (const float*)d_in[14], *ofe_b2 = (const float*)d_in[15];
  const float* ofe_beta = (const float*)d_in[16], *ofe_g = (const float*)d_in[17];
  const float* pfeW[4], *pfeB[4], *pfeBeta[4], *pfeG[4];
  for (int l = 0; l < 4; ++l) {
    pfeW[l]    = (const float*)d_in[18 + 4 * l];
    pfeB[l]    = (const float*)d_in[19 + 4 * l];
    pfeBeta[l] = (const float*)d_in[20 + 4 * l];
    pfeG[l]    = (const float*)d_in[21 + 4 * l];
  }
  const float* r1W1 = (const float*)d_in[34], *r1W2 = (const float*)d_in[35];
  const float* r1b1 = (const float*)d_in[36], *r1b2 = (const float*)d_in[37];
  const float* r1bn0beta = (const float*)d_in[38], *r1bn0g = (const float*)d_in[39];
  const float* r1bn1beta = (const float*)d_in[40], *r1bn1g = (const float*)d_in[41];
  const float* r1bn2beta = (const float*)d_in[42], *r1bn2g = (const float*)d_in[43];
  const float* r2W1 = (const float*)d_in[44], *r2W2 = (const float*)d_in[45];
  const float* r2b1 = (const float*)d_in[46], *r2b2 = (const float*)d_in[47];
  const float* r2bn1beta = (const float*)d_in[48], *r2bn1g = (const float*)d_in[49];
  const float* r2bn2beta = (const float*)d_in[50], *r2bn2g = (const float*)d_in[51];
  const float* sfeW[4], *sfeB[4], *sfeBeta[4], *sfeG[4];
  for (int l = 0; l < 4; ++l) {
    sfeW[l]    = (const float*)d_in[52 + 4 * l];
    sfeB[l]    = (const float*)d_in[53 + 4 * l];
    sfeBeta[l] = (const float*)d_in[54 + 4 * l];
    sfeG[l]    = (const float*)d_in[55 + 4 * l];
  }

  const int NS = 204800;           // B*S
  char* ws = (char*)d_ws;
  size_t off = 0;
  auto alloc = [&](size_t bytes) -> char* {
    char* p = ws + off;
    off += (bytes + 255) & ~(size_t)255;
    return p;
  };
  unsigned short* actA = (unsigned short*)alloc((size_t)256 * NS * 2);
  unsigned short* actB = (unsigned short*)alloc((size_t)256 * NS * 2);
  float* stats = (float*)alloc((size_t)18 * 1024 * 4);
  float* pfbn  = (float*)alloc(2800 * 4);
  unsigned short* Xpf = (unsigned short*)alloc(2800 * 2);
  unsigned short* pactA = (unsigned short*)alloc(256 * 200 * 2);
  unsigned short* pactB = (unsigned short*)alloc(256 * 200 * 2);
  float* dstf_red = (float*)alloc(51200 * 4);
  float* x0  = (float*)alloc((size_t)11 * 51200 * 4);
  float* y1  = (float*)alloc((size_t)64 * 51200 * 4);
  float* y2  = (float*)alloc((size_t)256 * 51200 * 4);
  float* den1 = (float*)alloc(1800 * 4);
  float* maxm = (float*)alloc(1800 * 4);
  float* avgrf1 = (float*)alloc((size_t)256 * 1800 * 4);
  float* y3  = (float*)alloc((size_t)256 * 1800 * 4);
  float* y4  = (float*)alloc((size_t)256 * 1800 * 4);
  float* avgrf2 = (float*)alloc(51200 * 4);
  float* pf1z   = (float*)alloc(51200 * 4);
  float* at2    = (float*)alloc(800 * 4);
  unsigned short* allf = (unsigned short*)alloc(51200 * 2);
  unsigned short* h5   = (unsigned short*)alloc(51200 * 2);
  // packed bf16 weights, zero-padded to Kp
  unsigned* wbSfe[4], *wbPfe[4];
  int sfeKl[4] = {15, 64, 256, 256}, sfeKp[4] = {32, 64, 256, 256}, sfeM[4] = {64, 256, 256, 256};
  int pfeKl[4] = {14, 64, 256, 256}, pfeKp[4] = {32, 64, 256, 256}, pfeM[4] = {64, 256, 256, 256};
  for (int l = 0; l < 4; ++l) wbSfe[l] = (unsigned*)alloc((size_t)sfeM[l] * (sfeKp[l] / 2) * 4);
  for (int l = 0; l < 4; ++l) wbPfe[l] = (unsigned*)alloc((size_t)pfeM[l] * (pfeKp[l] / 2) * 4);
  unsigned* wbOfe  = (unsigned*)alloc((size_t)256 * 128 * 4);
  unsigned* wbR1c1 = (unsigned*)alloc((size_t)64 * 64 * 4);      // Kp=128
  unsigned* wbR1c2 = (unsigned*)alloc((size_t)256 * 288 * 4);    // Kp=576
  unsigned* wbR2c1 = (unsigned*)alloc((size_t)256 * 1152 * 4);   // Kp=2304
  unsigned* wbR2c2 = (unsigned*)alloc((size_t)256 * 1152 * 4);
  float* out = (float*)d_out;

  auto slot = [&](int i) -> float* { return stats + (size_t)i * 1024; };
  float* dsum  = slot(16);
  float* nsum1 = slot(17);
  float* nsum2 = slot(17) + 1;
  const float EPS_BN = 1e-5f, EPS_SP = 1e-3f;
  auto wgrid = [](int cnt) { return (cnt + 255) / 256; };

  // 0) zero stat accumulators; pack all weights to bf16
  zero_f32<<<72, 256, 0, stream>>>(stats, 18 * 1024);
  for (int l = 0; l < 4; ++l) {
    int c = sfeM[l] * sfeKp[l] / 2;
    wprep<<<wgrid(c), 256, 0, stream>>>(sfeW[l], wbSfe[l], sfeM[l], sfeKl[l], sfeKp[l]);
    c = pfeM[l] * pfeKp[l] / 2;
    wprep<<<wgrid(c), 256, 0, stream>>>(pfeW[l], wbPfe[l], pfeM[l], pfeKl[l], pfeKp[l]);
  }
  wprep<<<wgrid(256 * 128), 256, 0, stream>>>(ofe_W1, wbOfe, 256, 256, 256);
  wprep<<<wgrid(64 * 64), 256, 0, stream>>>(r1W1, wbR1c1, 64, 99, 128);
  wprep<<<wgrid(256 * 288), 256, 0, stream>>>(r1W2, wbR1c2, 256, 576, 576);
  wprep<<<wgrid(256 * 1152), 256, 0, stream>>>(r2W1, wbR2c1, 256, 2304, 2304);
  wprep<<<wgrid(256 * 1152), 256, 0, stream>>>(r2W2, wbR2c2, 256, 2304, 2304);

  // ---- point-feature BN + affine ----
  pf_stats<<<14, 256, 0, stream>>>(pointfeature, slot(0));
  bn_finalize<<<1, 256, 0, stream>>>(slot(0), bnraw1_g, bnraw1_beta, 14, EPS_BN, 200.f, nullptr);
  pf_apply<<<11, 256, 0, stream>>>(pointfeature, slot(0), pfbn, Xpf);

  // ---- station branch ----
  dstf_build<<<800, 256, 0, stream>>>(stationfeat, stationpm, pfbn, actA, slot(1));
  smask_sum<<<200, 256, 0, stream>>>(stationmask, dsum);
  bn_finalize<<<1, 256, 0, stream>>>(slot(1), bnraw2_g, bnraw2_beta, 15, EPS_BN, (float)NS, nullptr);

  gemm_bn_t<1, true, false><<<dim3(1600, 2), 256, 0, stream>>>(
      wbSfe[0], sfeB[0], actA, slot(1) + 512, slot(1) + 768, actB, slot(6), 64, NS, 15, 32);
  bn_finalize<<<1, 256, 0, stream>>>(slot(6), sfeG[0], sfeBeta[0], 64, EPS_BN, (float)NS, nullptr);
  gemm_bn_t<2, true, true><<<dim3(1600, 8), 256, 0, stream>>>(
      wbSfe[1], sfeB[1], actB, slot(6) + 512, slot(6) + 768, actA, slot(7), 256, NS, 64, 64);
  bn_finalize<<<1, 256, 0, stream>>>(slot(7), sfeG[1], sfeBeta[1], 256, EPS_BN, (float)NS, nullptr);
  gemm_bn_t<2, true, true><<<dim3(1600, 8), 256, 0, stream>>>(
      wbSfe[2], sfeB[2], actA, slot(7) + 512, slot(7) + 768, actB, slot(8), 256, NS, 256, 256);
  bn_finalize<<<1, 256, 0, stream>>>(slot(8), sfeG[2], sfeBeta[2], 256, EPS_BN, (float)NS, nullptr);
  gemm_bn_t<2, true, true><<<dim3(1600, 8), 256, 0, stream>>>(
      wbSfe[3], sfeB[3], actB, slot(8) + 512, slot(8) + 768, actA, slot(9), 256, NS, 256, 256);
  bn_finalize<<<1, 256, 0, stream>>>(slot(9), sfeG[3], sfeBeta[3], 256, EPS_BN, (float)NS, nullptr);
  station_reduce<<<51200, 256, 0, stream>>>(actA, slot(9), stationmask, dsum, dstf_red);

  // ---- point MLP (N=200, tail-guarded) ----
  gemm_bn_t<0, false, false><<<dim3(2, 2), 256, 0, stream>>>(
      wbPfe[0], pfeB[0], Xpf, nullptr, nullptr, pactA, slot(2), 64, 200, 14, 32);
  bn_finalize<<<1, 256, 0, stream>>>(slot(2), pfeG[0], pfeBeta[0], 64, EPS_BN, 200.f, nullptr);
  gemm_bn_t<2, false, true><<<dim3(2, 8), 256, 0, stream>>>(
      wbPfe[1], pfeB[1], pactA, slot(2) + 512, slot(2) + 768, pactB, slot(3), 256, 200, 64, 64);
  bn_finalize<<<1, 256, 0, stream>>>(slot(3), pfeG[1], pfeBeta[1], 256, EPS_BN, 200.f, nullptr);
  gemm_bn_t<2, false, true><<<dim3(2, 8), 256, 0, stream>>>(
      wbPfe[2], pfeB[2], pactB, slot(3) + 512, slot(3) + 768, pactA, slot(4), 256, 200, 256, 256);
  bn_finalize<<<1, 256, 0, stream>>>(slot(4), pfeG[2], pfeBeta[2], 256, EPS_BN, 200.f, nullptr);
  gemm_bn_t<2, false, true><<<dim3(2, 8), 256, 0, stream>>>(
      wbPfe[3], pfeB[3], pactA, slot(4) + 512, slot(4) + 768, pactB, slot(5), 256, 200, 256, 256);
  bn_finalize<<<1, 256, 0, stream>>>(slot(5), pfeG[3], pfeBeta[3], 256, EPS_BN, 200.f, nullptr);
  pf1_head<<<200, 256, 0, stream>>>(pactB, slot(5), pf1z, at2);

  // ---- region branch ----
  mask_sum<<<200, 256, 0, stream>>>(regionmask, nsum1);
  x0_build<<<200, 256, 0, stream>>>(regionfeat, regionmask, x0, slot(10));
  bn_finalize<<<1, 256, 0, stream>>>(slot(10), r1bn0g, r1bn0beta, 11, EPS_SP, 0.f, nsum1);
  conv_gemm_t<1, true><<<dim3(400, 2), 256, 0, stream>>>(
      wbR1c1, r1b1, x0, slot(10) + 512, slot(10) + 768, regionmask, y1, slot(11), 64, 11, 51200, 16, 128);
  bn_finalize<<<1, 256, 0, stream>>>(slot(11), r1bn1g, r1bn1beta, 64, EPS_SP, 0.f, nsum1);
  conv_gemm_t<2, true><<<dim3(400, 8), 256, 0, stream>>>(
      wbR1c2, r1b2, y1, slot(11) + 512, slot(11) + 768, regionmask, y2, slot(12), 256, 64, 51200, 16, 576);
  bn_finalize<<<1, 256, 0, stream>>>(slot(12), r1bn2g, r1bn2beta, 256, EPS_SP, 0.f, nsum1);
  pool_mask<<<8, 256, 0, stream>>>(regionmask, den1, maxm, nsum2);
  pool_feat<<<1800, 256, 0, stream>>>(y2, regionmask, slot(12), den1, maxm, avgrf1);
  conv_gemm_t<0, false><<<dim3(15, 8), 256, 0, stream>>>(
      wbR2c1, r2b1, avgrf1, nullptr, nullptr, maxm, y3, slot(13), 256, 256, 1800, 3, 2304);
  bn_finalize<<<1, 256, 0, stream>>>(slot(13), r2bn1g, r2bn1beta, 256, EPS_SP, 0.f, nsum2);
  conv_gemm_t<2, false><<<dim3(15, 8), 256, 0, stream>>>(
      wbR2c2, r2b2, y3, slot(13) + 512, slot(13) + 768, maxm, y4, slot(14), 256, 256, 1800, 3, 2304);
  bn_finalize<<<1, 256, 0, stream>>>(slot(14), r2bn2g, r2bn2beta, 256, EPS_SP, 0.f, nsum2);
  final_pool<<<200, 256, 0, stream>>>(y4, slot(14), maxm, avgrf2);

  // ---- final head ----
  combine<<<200, 256, 0, stream>>>(pf1z, at2, cfe_W, cfe_b, dstf_red, avgrf2, allf);
  gemm_bn_t<0, false, true><<<dim3(2, 8), 256, 0, stream>>>(
      wbOfe, ofe_b1, allf, nullptr, nullptr, h5, slot(15), 256, 200, 256, 256);
  bn_finalize<<<1, 256, 0, stream>>>(slot(15), ofe_g, ofe_beta, 256, EPS_BN, 200.f, nullptr);
  out_head<<<200, 256, 0, stream>>>(h5, slot(15), ofe_W2, ofe_b2, out);
}
